// coords_update_38749194944929
// MI455X (gfx1250) — matrix operand order repaired and checked by reading, not compiled, not run on hardware
//
#include <hip/hip_runtime.h>
#include <hip/hip_fp16.h>

typedef __attribute__((ext_vector_type(16))) _Float16 v16h;
typedef __attribute__((ext_vector_type(8)))  float    v8f;
typedef __attribute__((ext_vector_type(4)))  float    v4f;   // native vector for nontemporal loads

#define E_EDGES 400000
#define H_HEADS 8
#define D_DIM   64
#define N_NODES 40000
#define G_GEN   10000

#define TOTAL_ROWS (E_EDGES * H_HEADS)   // 3,200,000 rows of [E*H, 64]
#define ROWS_PER_WAVE 16                 // one 16x16x32-f16 WMMA tile (x2 N-tiles, x2 K-chunks)
#define TOTAL_TILES (TOTAL_ROWS / ROWS_PER_WAVE)   // 200,000
#define WAVES_PER_BLOCK 8                // 256 threads = 8 wave32
#define TILES_PER_WAVE 20                // amortize W1-fragment build; 80KB contiguous A stream/wave
#define NUM_BLOCKS (TOTAL_TILES / (WAVES_PER_BLOCK * TILES_PER_WAVE))   // 1250, exact

// ---------------------------------------------------------------------------
// Kernel 0: zero the segment-sum accumulator agg[N,3] (workspace).
// ---------------------------------------------------------------------------
__global__ void zero_agg_kernel(float* __restrict__ agg, int n) {
    int i = blockIdx.x * blockDim.x + threadIdx.x;
    if (i < n) agg[i] = 0.0f;
}

// ---------------------------------------------------------------------------
// Kernel 1: fused  MLP(a_ij) -> attention[e] -> delta_x scatter-add.
//
// Each wave processes TILES_PER_WAVE consecutive 16-row tiles of [E*H,64]
// (one tile == 2 edges x 8 heads). Per tile:
//   C1 = A @ W1[:, 0:16],  C2 = A @ W1[:,16:32]   (4x v_wmma_f32_16x16x32_f16)
//   per-lane: acc = sum_r Wh[r]*leaky(C+b1).W2_cols  -> ONE 4-step half-wave
//   reduction; att = red + b2*sum(Wh). Lanes 0/16 finalize their edge:
//   normalize pos diff, scale by att, global_atomic_add_f32 into agg[src].
// B fragments (W1), b1/W2/Wh/b2/pro are built/loaded ONCE per wave.
// ---------------------------------------------------------------------------
__global__ __launch_bounds__(256) void attn_wmma_scatter_kernel(
    const float* __restrict__ a,       // [E,H,D] fp32 (streamed once, non-temporal)
    const float* __restrict__ pos,     // [N,3]
    const int*   __restrict__ ei,      // [2,E] int32
    const unsigned char* __restrict__ mask_inv, // [E] bool (1 byte)
    const int*   __restrict__ pro_ptr, // scalar pro_nodes_num
    const float* __restrict__ W1,      // [64,32] row-major
    const float* __restrict__ b1,      // [32]
    const float* __restrict__ W2,      // [32]
    const float* __restrict__ b2s,     // [1]
    const float* __restrict__ Wh,      // [8]
    float* __restrict__ agg)           // [N,3] accumulator
{
    const int lane   = threadIdx.x & 31;
    const int wave   = threadIdx.x >> 5;
    const int waveId = blockIdx.x * WAVES_PER_BLOCK + wave;   // wave-uniform
    const int tile0  = waveId * TILES_PER_WAVE;               // exact division, no remainder

    const int g   = lane >> 4;        // half-wave group (K-split of fragment layouts)
    const int nlo = lane & 15;        // row-within-tile / column-within-N-tile

    // ================= hoisted constants (once per wave) ===================
    // B 32x16 f16 layout: lane holds col N=nlo+16*t, K = c*32 + g*8 + j (+8 for j>=8)
    v16h bf[2][2];
    #pragma unroll
    for (int t = 0; t < 2; ++t) {
        #pragma unroll
        for (int c = 0; c < 2; ++c) {
            #pragma unroll
            for (int j = 0; j < 16; ++j) {
                const int K = c * 32 + g * 8 + j + (j >= 8 ? 8 : 0);
                bf[t][c][j] = (_Float16)W1[K * 32 + (nlo + 16 * t)];
            }
        }
    }
    const float b1lo = b1[nlo],  b1hi = b1[nlo + 16];
    const float w2lo = W2[nlo],  w2hi = W2[nlo + 16];
    const float b2   = b2s[0];
    float whr[8], sumWh = 0.0f;
    #pragma unroll
    for (int r = 0; r < 8; ++r) { whr[r] = Wh[r]; sumWh += whr[r]; }
    const float attBias = b2 * sumWh;    // b2 contribution folded in once
    const int pro = pro_ptr[0];

    // Per-lane A base: row nlo of tile0, K-group offset g*8.
    const float* rowPtr = a + ((long long)tile0 * ROWS_PER_WAVE + nlo) * D_DIM + g * 8;

    // ========================= tile loop ===================================
    for (int t = 0; t < TILES_PER_WAVE; ++t, rowPtr += ROWS_PER_WAVE * D_DIM) {
        // ---- A fragments: 16 rows x 64 f32, each float read exactly once --
        // chunk c: elems 0..7 : K = c*32 + g*8 + j ; elems 8..15: +16
        v16h af[2];
        #pragma unroll
        for (int c = 0; c < 2; ++c) {
            const v4f f0 = __builtin_nontemporal_load((const v4f*)(rowPtr + c * 32 + 0));
            const v4f f1 = __builtin_nontemporal_load((const v4f*)(rowPtr + c * 32 + 4));
            const v4f f2 = __builtin_nontemporal_load((const v4f*)(rowPtr + c * 32 + 16));
            const v4f f3 = __builtin_nontemporal_load((const v4f*)(rowPtr + c * 32 + 20));
            #pragma unroll
            for (int j = 0; j < 4; ++j) {
                af[c][j +  0] = (_Float16)f0[j];
                af[c][j +  4] = (_Float16)f1[j];
                af[c][j +  8] = (_Float16)f2[j];
                af[c][j + 12] = (_Float16)f3[j];
            }
        }

        // ---- 4x v_wmma_f32_16x16x32_f16 (2 N-tiles x 2 K-chunks) ----------
        v8f acc0 = {}; v8f acc1 = {};
        acc0 = __builtin_amdgcn_wmma_f32_16x16x32_f16(false, af[0], false, bf[0][0], (short)0, acc0, false, false);
        acc0 = __builtin_amdgcn_wmma_f32_16x16x32_f16(false, af[1], false, bf[0][1], (short)0, acc0, false, false);
        acc1 = __builtin_amdgcn_wmma_f32_16x16x32_f16(false, af[0], false, bf[1][0], (short)0, acc1, false, false);
        acc1 = __builtin_amdgcn_wmma_f32_16x16x32_f16(false, af[1], false, bf[1][1], (short)0, acc1, false, false);

        // ---- epilogue: fold W2 and Wh per lane, then ONE half-wave reduce -
        float local = 0.0f;
        #pragma unroll
        for (int r = 0; r < 8; ++r) {   // C row = r + 8*g ; head index = r
            float x0 = acc0[r] + b1lo;
            float x1 = acc1[r] + b1hi;
            x0 = (x0 > 0.0f) ? x0 : 0.01f * x0;          // leaky_relu(0.01)
            x1 = (x1 > 0.0f) ? x1 : 0.01f * x1;
            local += (x0 * w2lo + x1 * w2hi) * whr[r];
        }
        // reduce across the 16 lanes of this half-wave (masks <=8 stay in-half)
        local += __shfl_xor(local, 1);
        local += __shfl_xor(local, 2);
        local += __shfl_xor(local, 4);
        local += __shfl_xor(local, 8);
        const float att = local + attBias;

        // ---- per-edge finalize: 2 edges per tile (lane 0 and lane 16) -----
        if (nlo == 0) {
            const int e = (tile0 + t) * 2 + g;           // edge id
            const int src = ei[e];
            const int dst = ei[E_EDGES + e];
            const bool valid = (src >= pro) && (mask_inv[e] == 0);
            if (valid) {
                const float dx = pos[3 * src + 0] - pos[3 * dst + 0];
                const float dy = pos[3 * src + 1] - pos[3 * dst + 1];
                const float dz = pos[3 * src + 2] - pos[3 * dst + 2];
                const float nrm = sqrtf(dx * dx + dy * dy + dz * dz);
                const float s = att / (nrm + 1e-6f);
                atomicAdd(&agg[3 * src + 0], dx * s);
                atomicAdd(&agg[3 * src + 1], dy * s);
                atomicAdd(&agg[3 * src + 2], dz * s);
            }
        }
    }
}

// ---------------------------------------------------------------------------
// Kernel 2: out[i,:] = pos[gidx[i],:] + agg[gidx[i],:]
// ---------------------------------------------------------------------------
__global__ void gather_out_kernel(const float* __restrict__ pos,
                                  const float* __restrict__ agg,
                                  const int* __restrict__ gidx,
                                  float* __restrict__ out) {
    int i = blockIdx.x * blockDim.x + threadIdx.x;
    if (i < G_GEN) {
        const int n = gidx[i];
        out[3 * i + 0] = pos[3 * n + 0] + agg[3 * n + 0];
        out[3 * i + 1] = pos[3 * n + 1] + agg[3 * n + 1];
        out[3 * i + 2] = pos[3 * n + 2] + agg[3 * n + 2];
    }
}

// ---------------------------------------------------------------------------
// Launch. Input order (setup_inputs): a_ij, pos, generate_node_dist,
// edge_index, parent_node_idxes, generate_node_idxes, mask_edge_inv,
// pro_nodes_num, W1, b1, W2, b2, Wh.
// ---------------------------------------------------------------------------
extern "C" void kernel_launch(void* const* d_in, const int* in_sizes, int n_in,
                              void* d_out, int out_size, void* d_ws, size_t ws_size,
                              hipStream_t stream) {
    const float* a      = (const float*)d_in[0];
    const float* pos    = (const float*)d_in[1];
    // d_in[2] generate_node_dist: unused by reference
    const int*   ei     = (const int*)d_in[3];
    // d_in[4] parent_node_idxes: unused by reference
    const int*   gidx   = (const int*)d_in[5];
    const unsigned char* mask = (const unsigned char*)d_in[6];
    const int*   pro    = (const int*)d_in[7];
    const float* W1     = (const float*)d_in[8];
    const float* b1     = (const float*)d_in[9];
    const float* W2     = (const float*)d_in[10];
    const float* b2     = (const float*)d_in[11];
    const float* Wh     = (const float*)d_in[12];
    float* out = (float*)d_out;

    float* agg = (float*)d_ws;                 // N*3 floats = 480 KB scratch
    const int aggN = N_NODES * 3;

    zero_agg_kernel<<<(aggN + 255) / 256, 256, 0, stream>>>(agg, aggN);

    attn_wmma_scatter_kernel<<<NUM_BLOCKS, 256, 0, stream>>>(
        a, pos, ei, mask, pro, W1, b1, W2, b2, Wh, agg);

    gather_out_kernel<<<(G_GEN + 255) / 256, 256, 0, stream>>>(pos, agg, gidx, out);

    (void)in_sizes; (void)n_in; (void)out_size; (void)ws_size;
}
